// MultiHeadAttention_69896297775467
// MI455X (gfx1250) — compile-verified, matrix operand order checked by reading, not run on hardware
//
#include <hip/hip_runtime.h>
#include <hip/hip_bf16.h>

// MHA forward for MI455X (gfx1250, wave32, WMMA f32_16x16x32_f16 + TDM).
// B=2, S=2048, D=1024, H=16, HD=64.

typedef _Float16 h16 __attribute__((ext_vector_type(16)));
typedef _Float16 h8  __attribute__((ext_vector_type(8)));
typedef float    f8  __attribute__((ext_vector_type(8)));
typedef unsigned int u32x4 __attribute__((ext_vector_type(4)));
typedef int          i32x4 __attribute__((ext_vector_type(4)));
typedef int          i32x8 __attribute__((ext_vector_type(8)));

#define WMMA_F16(a, b, c) \
    __builtin_amdgcn_wmma_f32_16x16x32_f16(false, (a), false, (b), (short)0, (c), false, false)

// ---------------------------------------------------------------------------
// Fragment load, row-major f16 source, leading dim ld, tile origin (row0,col0).
// CDNA5 16-bit A-matrix 16x32 layout (ISA 7.12.2):
//   lane L: row = row0 + (L & 15)
//   element e: K = col0 + (e/8)*16 + (L/16)*8 + (e%8)
// -> two contiguous 8-half (16B) loads per lane. B^T fragments use the same
// mapping with lane = output column, so row-major W serves directly as B^T.
// Works for both global and LDS sources (addrspace inferred after inlining).
// ---------------------------------------------------------------------------
__device__ __forceinline__ h16 load_frag(const _Float16* __restrict__ p,
                                         int ld, int row0, int col0, int lane)
{
    const int r  = row0 + (lane & 15);
    const int kb = col0 + ((lane >> 4) << 3);
    const _Float16* q = p + (size_t)r * (size_t)ld + kb;
    h8 lo = *(const h8*)(q);        // K = kb + 0..7
    h8 hi = *(const h8*)(q + 16);   // K = kb + 16..23
    return __builtin_shufflevector(lo, hi, 0, 1, 2, 3, 4, 5, 6, 7,
                                           8, 9, 10, 11, 12, 13, 14, 15);
}

// ---------------------------------------------------------------------------
// Tensor Data Mover: async 2-D f16 tile load, global -> LDS.
// Descriptor per CDNA5 ISA §8.3/§8.4 (D# groups 0/1; groups 2/3 zero = 2-D).
//   tile_w elements per row (tile_w*2 bytes, multiple of 4B), tile_h rows,
//   row_stride = elements between consecutive tensor rows.
// Tracked with TENSORcnt; pair with __builtin_amdgcn_s_wait_tensorcnt.
// This toolchain exposes the 6-arg builtin: (g0, g1, g2, g3, i32x8, cpol).
// ---------------------------------------------------------------------------
__device__ __forceinline__ void tdm_load_2d_f16(const _Float16* gsrc,
                                                unsigned lds_addr,
                                                int tile_w, int tile_h,
                                                int row_stride)
{
    const unsigned long long ga = (unsigned long long)(size_t)gsrc;
    const unsigned td0 = (unsigned)row_stride;  // tensor_dim0 >= tile_w
    const unsigned td1 = (unsigned)tile_h;      // tensor_dim1 >= tile_h

    u32x4 g0;
    g0[0] = 1u;                                          // count=1, user D#
    g0[1] = lds_addr;                                    // LDS byte address
    g0[2] = (unsigned)(ga & 0xFFFFFFFFull);              // global_addr[31:0]
    g0[3] = (unsigned)((ga >> 32) & 0x1FFFFFFull)        // global_addr[56:32]
          | (2u << 30);                                  // type=2 ("image")

    i32x8 g1;
    g1[0] = (int)(1u << 16);                             // data_size=1 (2B)
    g1[1] = (int)((td0 & 0xFFFFu) << 16);                // tensor_dim0[15:0]
    g1[2] = (int)(((td0 >> 16) & 0xFFFFu)                // tensor_dim0[31:16]
          | ((td1 & 0xFFFFu) << 16));                    // tensor_dim1[15:0]
    g1[3] = (int)(((td1 >> 16) & 0xFFFFu)                // tensor_dim1[31:16]
          | (((unsigned)tile_w & 0xFFFFu) << 16));       // tile_dim0
    g1[4] = (int)((unsigned)tile_h & 0xFFFFu);           // tile_dim1; tile_dim2=0
    g1[5] = (int)(unsigned)row_stride;                   // tensor_dim0_stride[31:0]
    g1[6] = 0;                                           // stride hi, dim1_stride lo
    g1[7] = 0;                                           // dim1_stride hi

    const i32x4 gz4 = {};
    const i32x8 gz8 = {};
    __builtin_amdgcn_tensor_load_to_lds(g0, g1, gz4, gz4, gz8, 0);
}

// ---------------------------------------------------------------------------
// f32 -> f16 pack, 8 elements per thread (two b128 loads, one b128 store)
// ---------------------------------------------------------------------------
__global__ void pack_f16(const float* __restrict__ src,
                         _Float16* __restrict__ dst, int n8)
{
    int i = blockIdx.x * blockDim.x + threadIdx.x;
    if (i >= n8) return;
    const float4* s4 = (const float4*)src;
    float4 a = s4[2 * i];
    float4 b = s4[2 * i + 1];
    h8 o;
    o[0] = (_Float16)a.x; o[1] = (_Float16)a.y;
    o[2] = (_Float16)a.z; o[3] = (_Float16)a.w;
    o[4] = (_Float16)b.x; o[5] = (_Float16)b.y;
    o[6] = (_Float16)b.z; o[7] = (_Float16)b.w;
    *(h8*)(dst + (size_t)8 * i) = o;
}

// ---------------------------------------------------------------------------
// WMMA GEMM: Out[m,n] = sum_k A[m,k] * Bt[n,k]   (i.e. A @ Bt^T == x @ W.T)
// 256 threads = 8 waves; block owns a 128x64 output slab; each wave a 16x64
// tile. The shared 64x32 B-slab per k-step is staged into LDS by the TDM
// (double-buffered, issued by wave 0, synced with TENSORcnt + barrier); all
// 8 waves read B fragments from LDS (8x cut in B VMEM traffic). A fragments
// have no intra-block reuse and stay on direct global_load_b128.
// MODE 1: f16 out, per-head layout  [b, h, s, hd]      (Q, K)
// MODE 2: f16 out, transposed head  [b, h, hd, s]      (V)
// MODE 0: f16 out, row-major [m, n]                    (ctx)
// MODE 3: f32 out + bias, row-major                    (final projection)
// ---------------------------------------------------------------------------
template <int MODE>
__global__ __launch_bounds__(256) void gemm_wmma(
    const _Float16* __restrict__ A, const _Float16* __restrict__ Bt,
    void* __restrict__ Out, const float* __restrict__ bias,
    int M, int N, int K)
{
    __shared__ _Float16 bstage[2][64 * 32];   // 2 x 4 KiB double buffer

    const int lane = threadIdx.x & 31;
    const int wave = threadIdx.x >> 5;
    const int m0 = (blockIdx.x * 8 + wave) * 16;
    const int n0 = blockIdx.y * 64;
    const int ksteps = K / 32;

    if (wave == 0)      // prologue: stage B tile for k0 = 0
        tdm_load_2d_f16(Bt + (size_t)n0 * K,
                        (unsigned)(size_t)&bstage[0][0], 32, 64, K);

    f8 acc[4] = {};
    for (int t = 0; t < ksteps; ++t) {
        const int k0 = t * 32;
        if (wave == 0)
            __builtin_amdgcn_s_wait_tensorcnt(0);   // buf[t&1] resident
        __syncthreads();
        if (wave == 0 && t + 1 < ksteps)            // prefetch next slab
            tdm_load_2d_f16(Bt + (size_t)n0 * K + (k0 + 32),
                            (unsigned)(size_t)&bstage[(t + 1) & 1][0],
                            32, 64, K);

        const _Float16* pB = &bstage[t & 1][0];     // [64 rows][32 halves]
        h16 a = load_frag(A, K, m0, k0, lane);
#pragma unroll
        for (int i = 0; i < 4; ++i) {
            h16 b = load_frag(pB, 32, 16 * i, 0, lane);
            acc[i] = WMMA_F16(a, b, acc[i]);
        }
        __syncthreads();    // all waves done with buf[t&1] before overwrite
    }

    // C/D layout: lane L holds rows m = j + 8*(L/16), column n = L & 15
    const int nn = lane & 15;
    const int hf = lane >> 4;
#pragma unroll
    for (int i = 0; i < 4; ++i) {
        const int n = n0 + 16 * i + nn;
#pragma unroll
        for (int j = 0; j < 8; ++j) {
            const int m = m0 + j + 8 * hf;
            const float v = acc[i][j];
            if constexpr (MODE == 3) {
                ((float*)Out)[(size_t)m * N + n] = v + bias[n];
            } else if constexpr (MODE == 0) {
                ((_Float16*)Out)[(size_t)m * N + n] = (_Float16)v;
            } else if constexpr (MODE == 1) {
                const int b = m >> 11, s = m & 2047, h = n >> 6, d = n & 63;
                ((_Float16*)Out)[(((size_t)(b * 16 + h)) * 2048 + s) * 64 + d] =
                    (_Float16)v;
            } else { // MODE == 2: V transposed per head
                const int b = m >> 11, s = m & 2047, h = n >> 6, d = n & 63;
                ((_Float16*)Out)[(((size_t)(b * 16 + h)) * 64 + d) * 2048 + s] =
                    (_Float16)v;
            }
        }
    }
}

// ---------------------------------------------------------------------------
// Flash attention: one wave per 16-query tile, 32 keys per step.
// Qh, Kh: [32][2048][64] f16.  Vt: [32][64][2048] f16 (per-head transposed).
// Ctx out: [4096][1024] f16 (b,s major; col = h*64 + d).
// ---------------------------------------------------------------------------
__global__ __launch_bounds__(256) void attn_wmma(
    const _Float16* __restrict__ Qh, const _Float16* __restrict__ Kh,
    const _Float16* __restrict__ Vt, _Float16* __restrict__ Ctx)
{
    __shared__ _Float16 plds[8][16 * 48]; // 16 rows x 32 keys, stride 48 halves

    const int lane = threadIdx.x & 31;
    const int wave = threadIdx.x >> 5;
    const int gw = blockIdx.x * 8 + wave;   // 0..4095
    const int bh = gw >> 7;                 // 0..31  (b*16 + h)
    const int q0 = (gw & 127) * 16;

    const _Float16* Qb = Qh + (size_t)bh * 2048 * 64;
    const _Float16* Kb = Kh + (size_t)bh * 2048 * 64;
    const _Float16* Vb = Vt + (size_t)bh * 64 * 2048;

    const h16 qa0 = load_frag(Qb, 64, q0, 0, lane);
    const h16 qa1 = load_frag(Qb, 64, q0, 32, lane);

    f8 o0 = {}, o1 = {}, o2 = {}, o3 = {};
    float mrow[8], lrow[8];
#pragma unroll
    for (int j = 0; j < 8; ++j) { mrow[j] = -1e30f; lrow[j] = 0.0f; }

    const int nn = lane & 15;
    const int hf = lane >> 4;
    _Float16* pw = &plds[wave][0];
    const int nsteps = (q0 >> 5) + 1;

    for (int t = 0; t < nsteps; ++t) {
        const int k0 = t << 5;

        // ---- scores: S[16 q x 32 k] = Q(16x64) . K^T, 4 wmma -------------
        f8 s0 = {}, s1 = {};
        {
            h16 kb;
            kb = load_frag(Kb, 64, k0, 0, lane);       s0 = WMMA_F16(qa0, kb, s0);
            kb = load_frag(Kb, 64, k0, 32, lane);      s0 = WMMA_F16(qa1, kb, s0);
            kb = load_frag(Kb, 64, k0 + 16, 0, lane);  s1 = WMMA_F16(qa0, kb, s1);
            kb = load_frag(Kb, 64, k0 + 16, 32, lane); s1 = WMMA_F16(qa1, kb, s1);
        }

        // ---- scale + causal mask (only the diagonal step needs it) ------
        const bool last = (t == nsteps - 1);
        float tmax[8];
#pragma unroll
        for (int j = 0; j < 8; ++j) {
            float a = s0[j] * 0.125f;   // 1/sqrt(64)
            float b = s1[j] * 0.125f;
            if (last) {
                const int qrow = q0 + j + 8 * hf;
                if (k0 + nn > qrow)      a = -1e30f;
                if (k0 + 16 + nn > qrow) b = -1e30f;
            }
            s0[j] = a; s1[j] = b;
            tmax[j] = fmaxf(a, b);
        }

        // ---- online softmax (row reductions across 16-lane groups) ------
#pragma unroll
        for (int msk = 1; msk < 16; msk <<= 1)
#pragma unroll
            for (int j = 0; j < 8; ++j)
                tmax[j] = fmaxf(tmax[j], __shfl_xor(tmax[j], msk, 32));

        float scale[8], tsum[8];
#pragma unroll
        for (int j = 0; j < 8; ++j) {
            const float mn = fmaxf(mrow[j], tmax[j]);
            scale[j] = __expf(mrow[j] - mn);
            mrow[j] = mn;
            const float p0 = __expf(s0[j] - mn);
            const float p1 = __expf(s1[j] - mn);
            s0[j] = p0; s1[j] = p1;
            tsum[j] = p0 + p1;
        }
#pragma unroll
        for (int msk = 1; msk < 16; msk <<= 1)
#pragma unroll
            for (int j = 0; j < 8; ++j)
                tsum[j] += __shfl_xor(tsum[j], msk, 32);
#pragma unroll
        for (int j = 0; j < 8; ++j) {
            lrow[j] = lrow[j] * scale[j] + tsum[j];
            o0[j] *= scale[j]; o1[j] *= scale[j];
            o2[j] *= scale[j]; o3[j] *= scale[j];
        }

        // ---- P: C-layout -> A-layout transpose through LDS --------------
#pragma unroll
        for (int j = 0; j < 8; ++j) {
            const int r = j + 8 * hf;
            pw[r * 48 + nn]      = (_Float16)s0[j];
            pw[r * 48 + 16 + nn] = (_Float16)s1[j];
        }
        asm volatile("s_wait_dscnt 0x0" ::: "memory");
        const h16 pa = load_frag(pw, 48, 0, 0, lane);
        asm volatile("s_wait_dscnt 0x0" ::: "memory");

        // ---- O += P(16x32) . V(32x64), 4 wmma (Vt rows are contiguous) --
        h16 vb;
        vb = load_frag(Vb, 2048, 0,  k0, lane); o0 = WMMA_F16(pa, vb, o0);
        vb = load_frag(Vb, 2048, 16, k0, lane); o1 = WMMA_F16(pa, vb, o1);
        vb = load_frag(Vb, 2048, 32, k0, lane); o2 = WMMA_F16(pa, vb, o2);
        vb = load_frag(Vb, 2048, 48, k0, lane); o3 = WMMA_F16(pa, vb, o3);
    }

    // ---- epilogue: normalize, store ctx f16 [b, s, h*64 + d] ------------
    const int b = bh >> 4, h = bh & 15;
#pragma unroll
    for (int j = 0; j < 8; ++j) {
        const float inv = 1.0f / lrow[j];
        const int s = q0 + j + 8 * hf;
        _Float16* dst = Ctx + ((size_t)(b * 2048 + s)) * 1024 + h * 64;
        dst[nn]      = (_Float16)(o0[j] * inv);
        dst[16 + nn] = (_Float16)(o1[j] * inv);
        dst[32 + nn] = (_Float16)(o2[j] * inv);
        dst[48 + nn] = (_Float16)(o3[j] * inv);
    }
}

// ---------------------------------------------------------------------------
// Host orchestration
// ---------------------------------------------------------------------------
extern "C" void kernel_launch(void* const* d_in, const int* in_sizes, int n_in,
                              void* d_out, int out_size, void* d_ws, size_t ws_size,
                              hipStream_t stream)
{
    (void)in_sizes; (void)n_in; (void)out_size; (void)ws_size;

    const float* X  = (const float*)d_in[0];
    const float* Wq = (const float*)d_in[1];
    const float* Wk = (const float*)d_in[2];
    const float* Wv = (const float*)d_in[3];
    const float* Wo = (const float*)d_in[4];
    const float* bo = (const float*)d_in[5];

    // Workspace layout (bytes). Ctx reuses the Xh buffer (dead after V GEMM).
    char* ws = (char*)d_ws;
    _Float16* Xh  = (_Float16*)(ws);                    //  8 MiB [4096][1024]
    _Float16* Wqh = (_Float16*)(ws + (8u << 20));       //  2 MiB
    _Float16* Wkh = (_Float16*)(ws + (10u << 20));      //  2 MiB
    _Float16* Wvh = (_Float16*)(ws + (12u << 20));      //  2 MiB
    _Float16* Woh = (_Float16*)(ws + (14u << 20));      //  2 MiB
    _Float16* Qh  = (_Float16*)(ws + (16u << 20));      //  8 MiB [32][2048][64]
    _Float16* Kh  = (_Float16*)(ws + (24u << 20));      //  8 MiB [32][2048][64]
    _Float16* Vt  = (_Float16*)(ws + (32u << 20));      //  8 MiB [32][64][2048]
    _Float16* Ctx = Xh;                                 //  alias (8 MiB)

    // 1) pack f32 -> f16
    const int nX8 = (4096 * 1024) / 8;
    const int nW8 = (1024 * 1024) / 8;
    pack_f16<<<(nX8 + 255) / 256, 256, 0, stream>>>(X, Xh, nX8);
    pack_f16<<<(nW8 + 255) / 256, 256, 0, stream>>>(Wq, Wqh, nW8);
    pack_f16<<<(nW8 + 255) / 256, 256, 0, stream>>>(Wk, Wkh, nW8);
    pack_f16<<<(nW8 + 255) / 256, 256, 0, stream>>>(Wv, Wvh, nW8);
    pack_f16<<<(nW8 + 255) / 256, 256, 0, stream>>>(Wo, Woh, nW8);

    // 2) QKV projections (M=4096, N=1024, K=1024); grid = (M/128, N/64)
    const dim3 gg(32, 16);
    gemm_wmma<1><<<gg, 256, 0, stream>>>(Xh, Wqh, Qh, nullptr, 4096, 1024, 1024);
    gemm_wmma<1><<<gg, 256, 0, stream>>>(Xh, Wkh, Kh, nullptr, 4096, 1024, 1024);
    gemm_wmma<2><<<gg, 256, 0, stream>>>(Xh, Wvh, Vt, nullptr, 4096, 1024, 1024);

    // 3) causal flash attention: 4096 waves = 32 bh * 128 q-tiles
    attn_wmma<<<512, 256, 0, stream>>>(Qh, Kh, Vt, Ctx);

    // 4) output projection + bias, f32 result
    gemm_wmma<3><<<gg, 256, 0, stream>>>(Ctx, Woh, d_out, bo, 4096, 1024, 1024);
}
